// OntologyCommunityDetection_36438502539676
// MI455X (gfx1250) — compile-verified, matrix operand order checked by reading, not run on hardware
//
#include <hip/hip_runtime.h>
#include <hip/hip_fp16.h>

// CDNA5 / gfx1250, wave32. Matrix op: v_wmma_f32_16x16x32_f16 (f32 accum).
// Workspace requirement: ~88 MB (h 32.8MB, x_gat/x_enc 32.8MB, packed-A 16.4MB, alpha 4.2MB, misc).

typedef __attribute__((ext_vector_type(16))) _Float16 v16h;
typedef __attribute__((ext_vector_type(8)))  float    v8f;

#define CDIM 3   // communities (reference C=3)

// ---------------- helpers ----------------
__device__ __forceinline__ unsigned enc_f32(float f) {
  unsigned u = __float_as_uint(f);
  return (u & 0x80000000u) ? ~u : (u | 0x80000000u);   // order-preserving uint encoding
}
__device__ __forceinline__ float dec_f32(unsigned u) {
  return (u & 0x80000000u) ? __uint_as_float(u ^ 0x80000000u) : __uint_as_float(~u);
}
__device__ __forceinline__ void edge_sd(const int* __restrict__ ei, long long E,
                                        long long i, int& s, int& d) {
  if (i < E) { s = ei[i]; d = ei[E + i]; }
  else       { s = d = (int)(i - E); }    // appended self-loops
}

// ---------------- 1a) pack W_gat into per-lane f16 B-fragment layout ----------------
// B (32x16, KxN) lane layout (mirrors documented 16-bit A layout):
//   lane L: N = L%16, khalf = L/16; elem i: K = (i/8)*16 + khalf*8 + (i%8)
// stored as pw[((ct*ksCount + ks)*32 + lane)*16 + elem]
__global__ void pack_w_kernel(const float* __restrict__ W, _Float16* __restrict__ pw, int D) {
  int total = (D >> 4) * (D >> 5) * 32 * 16;
  int idx = blockIdx.x * blockDim.x + threadIdx.x;
  if (idx >= total) return;
  int ksCount = D >> 5;
  int elem = idx & 15;
  int lane = (idx >> 4) & 31;
  int grp  = idx >> 9;              // ct*ksCount + ks
  int ks   = grp % ksCount;
  int ct   = grp / ksCount;
  int k = ks * 32 + (elem >> 3) * 16 + (lane >> 4) * 8 + (elem & 7);
  int n = ct * 16 + (lane & 15);
  pw[idx] = (_Float16)W[(size_t)k * D + n];
}

// ---------------- 1b) pack x into per-lane f16 A-fragment layout ----------------
// A (16x32, MxK) documented 16-bit layout:
//   lane L: M = L%16, khalf = L/16; elem i: K = (i/8)*16 + khalf*8 + (i%8)
// stored as pa[((rt*ksCount + ks)*32 + lane)*16 + elem]  -> 32 contiguous bytes per lane
__global__ void pack_x_kernel(const float* __restrict__ x, _Float16* __restrict__ pa,
                              long long N, int D) {
  long long total = (N >> 4) * (long long)(D >> 5) * 512;
  long long idx = blockIdx.x * (long long)blockDim.x + threadIdx.x;
  if (idx >= total) return;
  int ksCount = D >> 5;
  int elem = (int)(idx & 15);
  int lane = (int)((idx >> 4) & 31);
  long long grp = idx >> 9;         // rt*ksCount + ks
  int ks = (int)(grp % ksCount);
  long long rt = grp / ksCount;
  long long row = rt * 16 + (lane & 15);
  int k = ks * 32 + (elem >> 3) * 16 + (lane >> 4) * 8 + (elem & 7);
  pa[idx] = (_Float16)x[row * D + k];
}

// ---------------- 2) h = x @ W_gat via WMMA: one wave per 16x64 stripe ----------------
// A fragment reused across 4 column tiles -> 4 WMMAs per A load.
__global__ void gat_gemm_wmma(const _Float16* __restrict__ pa, const _Float16* __restrict__ pw,
                              float* __restrict__ h, int N, int D) {
  int wid  = (int)((blockIdx.x * (long long)blockDim.x + threadIdx.x) >> 5);
  int lane = threadIdx.x & 31;
  int ctGroups = D >> 6;            // groups of 4 column tiles
  int rt  = wid / ctGroups;         // row tile (wave-uniform)
  int cg  = wid % ctGroups;
  int ct0 = cg * 4;
  if (rt * 16 >= N) return;         // wave-uniform exit: EXEC stays all-ones for WMMA
  int ksCount = D >> 5;
  const v16h* ap = (const v16h*)(pa + (((size_t)rt * ksCount) * 32 + lane) * 16);
  const v16h* bp = (const v16h*)(pw + (((size_t)ct0 * ksCount) * 32 + lane) * 16);
  v8f acc0 = {}, acc1 = {}, acc2 = {}, acc3 = {};
  for (int ks = 0; ks < ksCount; ++ks) {
    v16h a  = ap[ks * 32];                       // 32B contiguous per lane
    v16h b0 = bp[ks * 32];
    v16h b1 = bp[(ksCount + ks) * 32];
    v16h b2 = bp[(2 * ksCount + ks) * 32];
    v16h b3 = bp[(3 * ksCount + ks) * 32];
    acc0 = __builtin_amdgcn_wmma_f32_16x16x32_f16(false, a, false, b0, (short)0, acc0, false, false);
    acc1 = __builtin_amdgcn_wmma_f32_16x16x32_f16(false, a, false, b1, (short)0, acc1, false, false);
    acc2 = __builtin_amdgcn_wmma_f32_16x16x32_f16(false, a, false, b2, (short)0, acc2, false, false);
    acc3 = __builtin_amdgcn_wmma_f32_16x16x32_f16(false, a, false, b3, (short)0, acc3, false, false);
  }
  // D/C layout: VGPR v, lane L: M = v + 8*(L/16), N = L%16
  int mrow = rt * 16 + 8 * (lane >> 4);
  int nc   = (lane & 15);
  float* hb = h + (size_t)mrow * D + ct0 * 16 + nc;
  #pragma unroll
  for (int v = 0; v < 8; ++v) {
    size_t r = (size_t)v * D;
    hb[r]      = acc0[v];
    hb[r + 16] = acc1[v];
    hb[r + 32] = acc2[v];
    hb[r + 48] = acc3[v];
  }
}

// ---------------- 3) a_s = h@att_src, a_d = h@att_dst (one wave per node) ----------------
__global__ void att_dots_kernel(const float* __restrict__ h, const float* __restrict__ as_v,
                                const float* __restrict__ ad_v, float* __restrict__ a_s,
                                float* __restrict__ a_d, int N, int D) {
  int wid  = (int)((blockIdx.x * (long long)blockDim.x + threadIdx.x) >> 5);
  int lane = threadIdx.x & 31;
  if (wid >= N) return;
  const float* hr = h + (size_t)wid * D;
  float s = 0.f, d = 0.f;
  for (int i = lane; i < D; i += 32) { float v = hr[i]; s += v * as_v[i]; d += v * ad_v[i]; }
  #pragma unroll
  for (int m = 16; m; m >>= 1) { s += __shfl_xor(s, m, 32); d += __shfl_xor(d, m, 32); }
  if (lane == 0) { a_s[wid] = s; a_d[wid] = d; }
}

// ---------------- 4) edge logits + segment max ----------------
__global__ void edge_max_kernel(const int* __restrict__ ei, long long E, long long Etot,
                                const float* __restrict__ a_s, const float* __restrict__ a_d,
                                float* __restrict__ e, unsigned* __restrict__ menc) {
  long long i = blockIdx.x * (long long)blockDim.x + threadIdx.x;
  if (i >= Etot) return;
  int s, d; edge_sd(ei, E, i, s, d);
  float v = a_s[s] + a_d[d];
  v = (v > 0.f) ? v : 0.2f * v;           // leaky_relu(0.2)
  e[i] = v;
  atomicMax(&menc[d], enc_f32(v));
}

__global__ void decode_m_kernel(unsigned* __restrict__ m, int N) {
  int i = blockIdx.x * blockDim.x + threadIdx.x;
  if (i < N) { float f = dec_f32(m[i]); ((float*)m)[i] = f; }
}

// ---------------- 5) exp + segment sum ----------------
__global__ void edge_exp_kernel(const int* __restrict__ ei, long long E, long long Etot,
                                const float* __restrict__ m, float* __restrict__ e,
                                float* __restrict__ z) {
  long long i = blockIdx.x * (long long)blockDim.x + threadIdx.x;
  if (i >= Etot) return;
  int s, d; edge_sd(ei, E, i, s, d);
  float ex = __expf(e[i] - m[d]);
  e[i] = ex;
  atomicAdd(&z[d], ex);
}

// ---------------- 6) alpha + x_gat scatter (HEAVY: wave per edge) ----------------
__global__ void edge_gat_scatter(const int* __restrict__ ei, long long E, long long Etot,
                                 const float* __restrict__ z, const float* __restrict__ h,
                                 float* __restrict__ alpha, float* __restrict__ xgat,
                                 float* __restrict__ deg, int D) {
  long long wid = (blockIdx.x * (long long)blockDim.x + threadIdx.x) >> 5;
  int lane = threadIdx.x & 31;
  if (wid >= Etot) return;
  int s, d; edge_sd(ei, E, wid, s, d);
  float a = alpha[wid] / (z[d] + 1e-16f);     // all lanes read ex before lane0 overwrites
  if (lane == 0) { alpha[wid] = a; atomicAdd(&deg[d], a); }
  const float* hs = h + (size_t)s * D;
  float* xg = xgat + (size_t)d * D;
  for (int r = 0; r < (D >> 5); ++r) {
    int c = r * 32 + lane;                    // coalesced 128B load + coalesced atomics
    atomicAdd(&xg[c], a * hs[c]);
  }
}

// ---------------- 7) x_enc = elu(x_gat + b_gat); hg = x_enc @ W_gcn; deg -> dinv ----------------
__global__ void node_elu_hg(float* __restrict__ xenc, const float* __restrict__ bgat,
                            const float* __restrict__ Wgcn, float* __restrict__ hg,
                            float* __restrict__ deg, int N, int D) {
  int wid  = (int)((blockIdx.x * (long long)blockDim.x + threadIdx.x) >> 5);
  int lane = threadIdx.x & 31;
  if (wid >= N) return;
  float* xr = xenc + (size_t)wid * D;
  float c0 = 0.f, c1 = 0.f, c2 = 0.f;
  for (int r = 0; r < (D >> 5); ++r) {
    int c = r * 32 + lane;
    float v = xr[c] + bgat[c];
    v = (v > 0.f) ? v : expm1f(v);            // elu(alpha=1)
    xr[c] = v;
    c0 += v * Wgcn[c * CDIM + 0];
    c1 += v * Wgcn[c * CDIM + 1];
    c2 += v * Wgcn[c * CDIM + 2];
  }
  #pragma unroll
  for (int m = 16; m; m >>= 1) {
    c0 += __shfl_xor(c0, m, 32); c1 += __shfl_xor(c1, m, 32); c2 += __shfl_xor(c2, m, 32);
  }
  if (lane == 0) {
    hg[(size_t)wid * CDIM + 0] = c0;
    hg[(size_t)wid * CDIM + 1] = c1;
    hg[(size_t)wid * CDIM + 2] = c2;
    float dv = deg[wid];
    deg[wid] = (dv > 0.f) ? rsqrtf(dv) : 0.f; // in-place -> dinv
  }
}

// ---------------- 8) GCN scatter: x1[dst] += norm * hg[src] ----------------
__global__ void edge_gcn_scatter(const int* __restrict__ ei, long long E, long long Etot,
                                 const float* __restrict__ alpha, const float* __restrict__ dinv,
                                 const float* __restrict__ hg, float* __restrict__ x1) {
  long long i = blockIdx.x * (long long)blockDim.x + threadIdx.x;
  if (i >= Etot) return;
  int s, d; edge_sd(ei, E, i, s, d);
  float nrm = dinv[s] * alpha[i] * dinv[d];
  #pragma unroll
  for (int c = 0; c < CDIM; ++c)
    atomicAdd(&x1[(size_t)d * CDIM + c], nrm * hg[(size_t)s * CDIM + c]);
}

// ---------------- 9) leaky + LayerNorm + softmax -> comm (written to d_out) ----------------
__global__ void node_comm(const float* __restrict__ x1, const float* __restrict__ bgcn,
                          const float* __restrict__ g, const float* __restrict__ b,
                          float* __restrict__ out_comm, int N) {
  int i = blockIdx.x * blockDim.x + threadIdx.x;
  if (i >= N) return;
  float v[CDIM];
  #pragma unroll
  for (int c = 0; c < CDIM; ++c) {
    float t = x1[(size_t)i * CDIM + c] + bgcn[c];
    v[c] = (t > 0.f) ? t : 0.01f * t;         // leaky_relu(0.01)
  }
  float mu = (v[0] + v[1] + v[2]) * (1.f / CDIM);
  float var = 0.f;
  #pragma unroll
  for (int c = 0; c < CDIM; ++c) { float dd = v[c] - mu; var += dd * dd; }
  var *= (1.f / CDIM);
  float inv = rsqrtf(var + 1e-5f);
  float mx = -1e30f;
  #pragma unroll
  for (int c = 0; c < CDIM; ++c) { v[c] = (v[c] - mu) * inv * g[c] + b[c]; mx = fmaxf(mx, v[c]); }
  float ssum = 0.f;
  #pragma unroll
  for (int c = 0; c < CDIM; ++c) { v[c] = __expf(v[c] - mx); ssum += v[c]; }
  #pragma unroll
  for (int c = 0; c < CDIM; ++c) out_comm[(size_t)i * CDIM + c] = v[c] / ssum;
}

// ---------------- 10) fc2: logits[b,o] = x_enc_flat[b,:] . fc2_W[o,:] + bias ----------------
__global__ void fc2_kernel(const float* __restrict__ xenc, const float* __restrict__ W,
                           const float* __restrict__ bias, float* __restrict__ logits,
                           long long K, int OUT) {
  int b = blockIdx.x, o = blockIdx.y;
  const float* xr = xenc + (size_t)b * K;
  const float* wr = W + (size_t)o * K;
  float s = 0.f;
  for (long long k = (long long)threadIdx.x * 4; k < K; k += (long long)blockDim.x * 4) {
    __builtin_prefetch(wr + k + 4096, 0, 1);  // global_prefetch_b8
    float4 xv = *(const float4*)(xr + k);
    float4 wv = *(const float4*)(wr + k);
    s += xv.x * wv.x + xv.y * wv.y + xv.z * wv.z + xv.w * wv.w;
  }
  #pragma unroll
  for (int m = 16; m; m >>= 1) s += __shfl_xor(s, m, 32);
  __shared__ float sm[8];
  int lane = threadIdx.x & 31, wv = threadIdx.x >> 5;
  if (lane == 0) sm[wv] = s;
  __syncthreads();
  if (threadIdx.x == 0) {
    float t = 0.f;
    int nw = blockDim.x >> 5;
    for (int w = 0; w < nw; ++w) t += sm[w];
    logits[b * OUT + o] = t + bias[o];
  }
}

// ---------------- 11) softmax over OUT -> probs (d_out[0 .. B*OUT)) ----------------
__global__ void probs_kernel(const float* __restrict__ logits, float* __restrict__ out,
                             int B, int OUT) {
  int b = blockIdx.x * blockDim.x + threadIdx.x;
  if (b >= B) return;
  float mx = -1e30f;
  for (int o = 0; o < OUT; ++o) mx = fmaxf(mx, logits[b * OUT + o]);
  float ex[16], ssum = 0.f;
  for (int o = 0; o < OUT; ++o) { ex[o] = __expf(logits[b * OUT + o] - mx); ssum += ex[o]; }
  for (int o = 0; o < OUT; ++o) out[b * OUT + o] = ex[o] / ssum;
}

// =====================================================================
extern "C" void kernel_launch(void* const* d_in, const int* in_sizes, int n_in,
                              void* d_out, int out_size, void* d_ws, size_t ws_size,
                              hipStream_t stream) {
  const float* x     = (const float*)d_in[0];
  const int*   ei    = (const int*)d_in[1];
  const float* W_gat = (const float*)d_in[2];
  const float* att_s = (const float*)d_in[3];
  const float* att_d = (const float*)d_in[4];
  const float* b_gat = (const float*)d_in[5];
  const float* W_gcn = (const float*)d_in[6];
  const float* b_gcn = (const float*)d_in[7];
  const float* ln_g  = (const float*)d_in[8];
  const float* ln_b  = (const float*)d_in[9];
  const float* fc2_W = (const float*)d_in[10];
  const float* fc2_b = (const float*)d_in[11];

  const int       D    = in_sizes[3];          // 256
  const long long N    = in_sizes[0] / D;      // 32000
  const long long E    = in_sizes[1] / 2;      // 1,024,000
  const long long Etot = E + N;                // + self loops
  const int       OUT  = in_sizes[11];         // 8
  const long long K    = (long long)in_sizes[10] / OUT;  // 512000
  const int       B    = (int)((N * D) / K);   // 16

  // ---- workspace carve (256B aligned) ----
  char* p = (char*)d_ws;
  auto carve = [&](size_t bytes) { char* r = p; p += (bytes + 255) & ~(size_t)255; return r; };
  float*    h      = (float*)   carve((size_t)N * D * 4);
  _Float16* pw     = (_Float16*)carve((size_t)(D / 16) * (D / 32) * 32 * 16 * 2);
  _Float16* pa     = (_Float16*)carve((size_t)N * D * 2);
  float*    a_s    = (float*)   carve((size_t)N * 4);
  float*    a_d    = (float*)   carve((size_t)N * 4);
  unsigned* menc   = (unsigned*)carve((size_t)N * 4);
  float*    ebuf   = (float*)   carve((size_t)Etot * 4);    // e -> ex -> alpha in place
  float*    z      = (float*)   carve((size_t)N * 4);
  float*    xgat   = (float*)   carve((size_t)N * D * 4);   // x_gat -> x_enc in place
  float*    deg    = (float*)   carve((size_t)N * 4);       // deg -> dinv in place
  float*    hg     = (float*)   carve((size_t)N * CDIM * 4);
  float*    x1     = (float*)   carve((size_t)N * CDIM * 4);
  float*    logits = (float*)   carve((size_t)B * OUT * 4);
  (void)ws_size; (void)n_in;

  float* out_probs = (float*)d_out;
  float* out_comm  = (float*)d_out + (size_t)B * OUT;

  // ---- zero accumulators (capturable stream memsets) ----
  hipMemsetAsync(menc, 0, (size_t)N * 4, stream);   // enc(-inf) < 0x00800000, 0 is a safe identity
  hipMemsetAsync(z,    0, (size_t)N * 4, stream);
  hipMemsetAsync(xgat, 0, (size_t)N * D * 4, stream);
  hipMemsetAsync(deg,  0, (size_t)N * 4, stream);
  hipMemsetAsync(x1,   0, (size_t)N * CDIM * 4, stream);

  const int BS = 256;
  // 1) pack W and x into WMMA fragment layouts (f32 -> f16 done once, off the hot loop)
  int packWTotal = (D / 16) * (D / 32) * 32 * 16;
  pack_w_kernel<<<(packWTotal + BS - 1) / BS, BS, 0, stream>>>(W_gat, pw, D);
  long long packXTotal = (N / 16) * (long long)(D / 32) * 512;
  pack_x_kernel<<<(unsigned)((packXTotal + BS - 1) / BS), BS, 0, stream>>>(x, pa, N, D);
  // 2) GEMM: one wave per 16x64 stripe (A fragment reused across 4 column tiles)
  long long gemmThreads = (N / 16) * (long long)(D / 64) * 32;
  gat_gemm_wmma<<<(unsigned)((gemmThreads + BS - 1) / BS), BS, 0, stream>>>(pa, pw, h, (int)N, D);
  // 3) attention dot products: one wave per node
  long long nodeWThreads = N * 32;
  att_dots_kernel<<<(unsigned)((nodeWThreads + BS - 1) / BS), BS, 0, stream>>>(
      h, att_s, att_d, a_s, a_d, (int)N, D);
  // 4) edge logits + segment max
  unsigned edgeBlocks = (unsigned)((Etot + BS - 1) / BS);
  edge_max_kernel<<<edgeBlocks, BS, 0, stream>>>(ei, E, Etot, a_s, a_d, ebuf, menc);
  decode_m_kernel<<<(unsigned)((N + BS - 1) / BS), BS, 0, stream>>>(menc, (int)N);
  // 5) exp + segment sum
  edge_exp_kernel<<<edgeBlocks, BS, 0, stream>>>(ei, E, Etot, (const float*)menc, ebuf, z);
  // 6) alpha + x_gat scatter (wave per edge)
  long long scatterThreads = Etot * 32;
  edge_gat_scatter<<<(unsigned)((scatterThreads + BS - 1) / BS), BS, 0, stream>>>(
      ei, E, Etot, z, h, ebuf, xgat, deg, D);
  // 7) elu + hg + dinv
  node_elu_hg<<<(unsigned)((nodeWThreads + BS - 1) / BS), BS, 0, stream>>>(
      xgat, b_gat, W_gcn, hg, deg, (int)N, D);
  // 8) GCN scatter
  edge_gcn_scatter<<<edgeBlocks, BS, 0, stream>>>(ei, E, Etot, ebuf, deg, hg, x1);
  // 9) LN + softmax -> comm (second output region)
  node_comm<<<(unsigned)((N + BS - 1) / BS), BS, 0, stream>>>(x1, b_gcn, ln_g, ln_b,
                                                             out_comm, (int)N);
  // 10) fc2 logits
  dim3 fcGrid((unsigned)B, (unsigned)OUT);
  fc2_kernel<<<fcGrid, BS, 0, stream>>>(xgat /*== x_enc*/, fc2_W, fc2_b, logits, K, OUT);
  // 11) probs softmax (first output region)
  probs_kernel<<<1, 64, 0, stream>>>(logits, out_probs, B, OUT);
}
